// CausalAttention_37452114821625
// MI455X (gfx1250) — compile-verified, matrix operand order checked by reading, not run on hardware
//
#include <hip/hip_runtime.h>

// MI455X (gfx1250) causal attention, FP16 WMMA path.
// ~120 GFLOP vs ~100 MB traffic (23.3 TB/s -> ~4us floor) => compute bound:
// everything runs through v_wmma_f32_16x16x32_f16 with f32 accumulation.

typedef __attribute__((ext_vector_type(16))) _Float16 v16h;
typedef __attribute__((ext_vector_type(8)))  _Float16 v8h;
typedef __attribute__((ext_vector_type(8)))  float    v8f;
typedef __attribute__((ext_vector_type(4)))  unsigned int v4u;
typedef __attribute__((ext_vector_type(8)))  int v8i;
typedef __attribute__((ext_vector_type(4)))  int v4i;

constexpr int kD = 1024;   // D_IN == D_OUT
constexpr int kS = 2048;   // SEQ
constexpr int kB = 4;      // BATCH

#if defined(__has_builtin)
#if __has_builtin(__builtin_amdgcn_tensor_load_to_lds) && \
    __has_builtin(__builtin_amdgcn_s_wait_tensorcnt)
#define HAVE_TDM 1
#endif
#endif

// ---------------------------------------------------------------------------
// WMMA helpers
// ---------------------------------------------------------------------------
__device__ __forceinline__ v8f wmma_f16(v16h a, v16h b, v8f c) {
  // (neg_a, A, neg_b, B, c_mod, C, reuse_a, reuse_b)
  return __builtin_amdgcn_wmma_f32_16x16x32_f16(false, a, false, b,
                                                (short)0, c, false, false);
}

// Load one 16x32 f16 fragment (A layout; B uses the same per-lane pattern with
// "row" == column index) from a row-major tile with leading dimension `ld`
// (in halves). Per ISA 7.12.2: lanes 0-15 hold K=[0..7],[16..23] of row m=lane,
// lanes 16-31 hold K=[8..15],[24..31] of row m=lane-16. Two 16B loads per lane.
__device__ __forceinline__ v16h load_frag(const _Float16* tile, int ld) {
  const int lane = threadIdx.x & 31;
  const _Float16* p = tile + (size_t)(lane & 15) * ld + ((lane >> 4) << 3);
  v8h lo = *(const v8h*)(p);        // K = koff + 0..7
  v8h hi = *(const v8h*)(p + 16);   // K = koff + 16..23
  return __builtin_shufflevector(lo, hi, 0, 1, 2, 3, 4, 5, 6, 7,
                                         8, 9, 10, 11, 12, 13, 14, 15);
}

#ifdef HAVE_TDM
// Tensor Data Mover: async-DMA a 2D f16 tile (rows x row_elems, row-major with
// stride row_stride_elems) from global memory into LDS at lds_off.
// Descriptor layout per CDNA5 ISA ch.8 (D# group0/group1).
__device__ __forceinline__ void tdm_load_2d_f16(unsigned int lds_off,
                                                const void* gptr, int rows,
                                                int row_elems,
                                                int row_stride_elems) {
  const unsigned long long ga = (unsigned long long)(uintptr_t)gptr;
  v4u g0 = {
      1u,                                         // count=1, user D#
      lds_off,                                    // lds_addr [63:32]
      (unsigned int)(ga & 0xFFFFFFFFu),           // global_addr [95:64]
      (unsigned int)((ga >> 32) & 0x01FFFFFFu) |  // global_addr [120:96]
          (2u << 30)                              // type=2 (image) [127:126]
  };
  v8i g1 = {
      (int)(1u << 16),                                    // data_size=1 (2B)
      (int)(((unsigned)row_elems & 0xFFFFu) << 16),       // tensor_dim0 lo
      (int)((((unsigned)row_elems >> 16) & 0xFFFFu) |
            (((unsigned)rows & 0xFFFFu) << 16)),          // dim0 hi | dim1 lo
      (int)(((unsigned)row_elems & 0xFFFFu) << 16),       // dim1 hi=0|tile_dim0
      (int)((unsigned)rows & 0xFFFFu),                    // tile_dim1|tile_dim2=0
      (int)row_stride_elems,                              // dim0_stride lo
      0,                                                  // stride hi | dim1_stride lo
      0
  };
  v4i z = {0, 0, 0, 0};
#if __clang_major__ >= 23
  v8i z8 = {0, 0, 0, 0, 0, 0, 0, 0};
  __builtin_amdgcn_tensor_load_to_lds(g0, g1, z, z, z8, 0);
#else
  __builtin_amdgcn_tensor_load_to_lds(g0, g1, z, z, 0);
#endif
}
#endif

// ---------------------------------------------------------------------------
// Prep kernels: f32 -> f16 convert, and weight transpose (W[k][n] -> Wt[n][k])
// ---------------------------------------------------------------------------
__global__ void cvt_f32_to_f16(const float* __restrict__ in,
                               _Float16* __restrict__ out, int n) {
  for (int i = blockIdx.x * blockDim.x + threadIdx.x; i < n;
       i += gridDim.x * blockDim.x)
    out[i] = (_Float16)in[i];
}

__global__ void transpose_w(const float* __restrict__ W,
                            _Float16* __restrict__ Wt) {
  int i = blockIdx.x * blockDim.x + threadIdx.x;  // over kD*kD
  if (i < kD * kD) {
    int n = i >> 10;
    int k = i & (kD - 1);
    Wt[i] = (_Float16)W[(size_t)k * kD + n];
  }
}

// ---------------------------------------------------------------------------
// Projection GEMM: Out[M=8192, N=1024] = A[M,K=1024] * Wt[N,K]^T  (f16 in/out,
// f32 accumulate). 8 waves/block; each wave computes a 64x64 macro-tile
// (4x4 grid of 16x16 WMMA tiles -> 16 b128 loads per 16 WMMAs).
// Block tile = 128 (M) x 256 (N).
// TRANS_OUT=1 writes Out transposed per batch: Vt[b][n][s] (for P*V later).
// ---------------------------------------------------------------------------
template <int TRANS_OUT>
__global__ __launch_bounds__(256) void qkv_gemm(
    const _Float16* __restrict__ A, const _Float16* __restrict__ Wt,
    _Float16* __restrict__ Out) {
  const int lane = threadIdx.x & 31;
  const int wave = threadIdx.x >> 5;
  const int col0 = blockIdx.x * 256 + (wave & 3) * 64;
  const int row0 = blockIdx.y * 128 + (wave >> 2) * 64;

  v8f zero = {};
  v8f c[4][4];
#pragma unroll
  for (int i = 0; i < 4; ++i)
#pragma unroll
    for (int j = 0; j < 4; ++j) c[i][j] = zero;

  for (int k0 = 0; k0 < kD; k0 += 32) {
    v16h a[4], b[4];
#pragma unroll
    for (int i = 0; i < 4; ++i)
      a[i] = load_frag(A + (size_t)(row0 + i * 16) * kD + k0, kD);
#pragma unroll
    for (int j = 0; j < 4; ++j)
      b[j] = load_frag(Wt + (size_t)(col0 + j * 16) * kD + k0, kD);
#pragma unroll
    for (int i = 0; i < 4; ++i)
#pragma unroll
      for (int j = 0; j < 4; ++j) c[i][j] = wmma_f16(a[i], b[j], c[i][j]);
  }

  // C/D layout: lane holds column n = lane&15; VGPR r holds row m = r + 8*(lane>=16)
  const int n = lane & 15;
#pragma unroll
  for (int i = 0; i < 4; ++i) {
#pragma unroll
    for (int j = 0; j < 4; ++j) {
#pragma unroll
      for (int r = 0; r < 8; ++r) {
        const int m = r + ((lane >> 4) << 3);
        const int grow = row0 + i * 16 + m;   // flattened b*S + s
        const int gcol = col0 + j * 16 + n;   // channel
        const _Float16 v = (_Float16)c[i][j][r];
        if (TRANS_OUT) {
          const int bi = grow >> 11;          // / kS
          const int s = grow & (kS - 1);
          Out[((size_t)bi * kD + gcol) * kS + s] = v;
        } else {
          Out[(size_t)grow * kD + gcol] = v;
        }
      }
    }
  }
}

// ---------------------------------------------------------------------------
// Flash attention (causal, single head, d=1024).
// One workgroup (8 waves / 256 threads) per (batch, 16-query tile).
// Per key block of 128 keys:
//   - wave w computes 16x16 score tile vs keys [kb*128+16w, +16) : 32 WMMAs
//   - online softmax: each wave owns 2 rows, 16-lane shfl_xor reductions
//   - wave w owns output columns [128w, 128w+128): O += P * Vt : 32 WMMAs
// Q tile staged in LDS via the Tensor Data Mover (TDM) when available.
// Vt is pre-transposed so its B-fragments are contiguous global b128 loads.
// ---------------------------------------------------------------------------
__global__ __launch_bounds__(256) void flash_attn(
    const _Float16* __restrict__ Q, const _Float16* __restrict__ K,
    const _Float16* __restrict__ Vt, float* __restrict__ Out) {
  __shared__ __align__(16) _Float16 q_lds[16 * kD];    // 32 KB
  __shared__ __align__(16) float    s_lds[16 * 128];   // 8 KB raw scores
  __shared__ __align__(16) _Float16 p_lds[16 * 128];   // 4 KB probabilities
  __shared__ float row_m[16], row_l[16], row_scale[16];

  const int tid  = threadIdx.x;
  const int lane = tid & 31;
  const int wave = tid >> 5;
  const int batch = blockIdx.x >> 7;          // 128 query tiles per batch
  const int qt = blockIdx.x & 127;
  const int q0 = qt << 4;

  const _Float16* Qg = Q + ((size_t)(batch * kS + q0)) * kD;
  const _Float16* Kg = K + (size_t)(batch * kS) * kD;
  const _Float16* Vg = Vt + (size_t)batch * kD * kS;

  // Stage the 16x1024 Q tile into LDS (read 16x per key block by every wave).
#ifdef HAVE_TDM
  if (wave == 0) {
    tdm_load_2d_f16((unsigned int)(uintptr_t)q_lds, Qg, 16, kD, kD);
    __builtin_amdgcn_s_wait_tensorcnt(0);
  }
#else
  {
    const v8h* src = (const v8h*)Qg;
    v8h* dst = (v8h*)q_lds;
    for (int i = tid; i < (16 * kD) / 8; i += 256) dst[i] = src[i];
  }
#endif
  if (tid < 16) { row_m[tid] = -1e30f; row_l[tid] = 0.f; }
  __syncthreads();

  v8f zero = {};
  v8f o[8];
#pragma unroll
  for (int j = 0; j < 8; ++j) o[j] = zero;

  const float sm_scale = 0.03125f;            // 1/sqrt(1024)
  const int nkb = (q0 + 15) / 128 + 1;        // causal: only key blocks <= q

  for (int kb = 0; kb < nkb; ++kb) {
    const int key0 = kb * 128;

    // Prefetch a slice of the next key block's K rows (global_prefetch_b8).
    if (kb + 1 < nkb) {
      const _Float16* Kn = Kg + (size_t)(key0 + 128 + wave * 16) * kD;
      __builtin_prefetch(Kn + (size_t)lane * (kD / 2), 0, 3);
    }

    // --- scores: this wave's 16 keys, full d=1024 reduction ---
    v8f sc = zero;
    const _Float16* Kw = Kg + (size_t)(key0 + wave * 16) * kD;
    for (int k0 = 0; k0 < kD; k0 += 32) {
      v16h a = load_frag(q_lds + k0, kD);     // LDS A-fragment
      v16h b = load_frag(Kw + k0, kD);        // global B-fragment (row-major K)
      sc = wmma_f16(a, b, sc);
    }

    // --- scale + causal mask, spill to LDS ---
    {
      const int n = lane & 15;
      const int col = wave * 16 + n;
      const int key = key0 + col;
#pragma unroll
      for (int r = 0; r < 8; ++r) {
        const int m = r + ((lane >> 4) << 3);
        const float sv = (key <= q0 + m) ? sc[r] * sm_scale : -1e30f;
        s_lds[m * 128 + col] = sv;
      }
    }
    __syncthreads();

    // --- online softmax: wave w owns rows {2w, 2w+1}; 16 lanes per row,
    //     8 columns per lane, shfl_xor tree across the 16-lane group ---
    {
      const int r = 2 * wave + (lane >> 4);
      const int c0 = (lane & 15) * 8;
      const float* srow = s_lds + r * 128;
      float mx = -1e30f;
#pragma unroll
      for (int c = 0; c < 8; ++c) mx = fmaxf(mx, srow[c0 + c]);
#pragma unroll
      for (int off = 1; off < 16; off <<= 1)
        mx = fmaxf(mx, __shfl_xor(mx, off, 32));
      mx = fmaxf(mx, row_m[r]);
      float sum = 0.f;
      _Float16* prow = p_lds + r * 128;
#pragma unroll
      for (int c = 0; c < 8; ++c) {
        const float p = __expf(srow[c0 + c] - mx);
        prow[c0 + c] = (_Float16)p;
        sum += p;
      }
#pragma unroll
      for (int off = 1; off < 16; off <<= 1) sum += __shfl_xor(sum, off, 32);
      if ((lane & 15) == 0) {
        const float sf = __expf(row_m[r] - mx);
        row_l[r] = row_l[r] * sf + sum;
        row_scale[r] = sf;
        row_m[r] = mx;
      }
    }
    __syncthreads();

    // --- rescale accumulators ---
#pragma unroll
    for (int r = 0; r < 8; ++r) {
      const float f = row_scale[r + ((lane >> 4) << 3)];
#pragma unroll
      for (int j = 0; j < 8; ++j) o[j][r] *= f;
    }

    // --- O += P * V  (wave's 128-column slice) ---
    v16h pa[4];
#pragma unroll
    for (int t = 0; t < 4; ++t) pa[t] = load_frag(p_lds + t * 32, 128);
    const _Float16* Vw = Vg + (size_t)(wave * 128) * kS + key0;
#pragma unroll
    for (int j = 0; j < 8; ++j) {
#pragma unroll
      for (int t = 0; t < 4; ++t) {
        v16h bv = load_frag(Vw + (size_t)(j * 16) * kS + t * 32, kS);
        o[j] = wmma_f16(pa[t], bv, o[j]);
      }
    }
    __syncthreads();  // protect p_lds/s_lds before next iteration
  }

  // --- epilogue: divide by softmax sum, store f32 ---
  {
    const int n = lane & 15;
#pragma unroll
    for (int r = 0; r < 8; ++r) {
      const int m = r + ((lane >> 4) << 3);
      const float inv = 1.0f / row_l[m];
      const size_t row = (size_t)(batch * kS + q0 + m) * kD;
#pragma unroll
      for (int j = 0; j < 8; ++j)
        Out[row + wave * 128 + j * 16 + n] = o[j][r] * inv;
    }
  }
}

// ---------------------------------------------------------------------------
// Launch
// ---------------------------------------------------------------------------
extern "C" void kernel_launch(void* const* d_in, const int* in_sizes, int n_in,
                              void* d_out, int out_size, void* d_ws,
                              size_t ws_size, hipStream_t stream) {
  (void)in_sizes; (void)n_in; (void)out_size; (void)ws_size;
  const float* x  = (const float*)d_in[0];
  const float* Wq = (const float*)d_in[1];
  const float* Wk = (const float*)d_in[2];
  const float* Wv = (const float*)d_in[3];
  float* out = (float*)d_out;

  char* ws = (char*)d_ws;
  _Float16* x16 = (_Float16*)ws; ws += (size_t)kB * kS * kD * 2;  // 16 MB
  _Float16* wqt = (_Float16*)ws; ws += (size_t)kD * kD * 2;       // 2 MB
  _Float16* wkt = (_Float16*)ws; ws += (size_t)kD * kD * 2;
  _Float16* wvt = (_Float16*)ws; ws += (size_t)kD * kD * 2;
  _Float16* Qb  = (_Float16*)ws; ws += (size_t)kB * kS * kD * 2;  // 16 MB
  _Float16* Kb  = (_Float16*)ws; ws += (size_t)kB * kS * kD * 2;  // 16 MB
  _Float16* Vt  = (_Float16*)ws; ws += (size_t)kB * kS * kD * 2;  // 16 MB

  cvt_f32_to_f16<<<2048, 256, 0, stream>>>(x, x16, kB * kS * kD);
  transpose_w<<<(kD * kD + 255) / 256, 256, 0, stream>>>(Wq, wqt);
  transpose_w<<<(kD * kD + 255) / 256, 256, 0, stream>>>(Wk, wkt);
  transpose_w<<<(kD * kD + 255) / 256, 256, 0, stream>>>(Wv, wvt);

  dim3 ggrid(kD / 256, kB * kS / 128);  // (4, 64)
  qkv_gemm<0><<<ggrid, 256, 0, stream>>>(x16, wqt, Qb);
  qkv_gemm<0><<<ggrid, 256, 0, stream>>>(x16, wkt, Kb);
  qkv_gemm<1><<<ggrid, 256, 0, stream>>>(x16, wvt, Vt);

  flash_attn<<<kB * kS / 16, 256, 0, stream>>>(Qb, Kb, Vt, out);
}
